// LSTM_39599598469783
// MI455X (gfx1250) — compile-verified, matrix operand order checked by reading, not run on hardware
//
#include <hip/hip_runtime.h>
#include <math.h>

// ---------------- problem constants ----------------
constexpr int BATCH  = 512;
constexpr int INPUT  = 128;
constexpr int SEQ    = 256;
constexpr int HIDDEN = 1024;
constexpr int OUTF   = 128;
constexpr int G4     = 4 * HIDDEN;        // 4096 gate rows
constexpr int KC     = INPUT + HIDDEN;    // 1152 combined K (x | h)
constexpr int CK     = 64;                // K-chunk staged in LDS
constexpr int NCHUNK = KC / CK;           // 18

// ---------------- workspace layout (bytes, all 16B aligned) ----------------
constexpr size_t WS_WC    = 0;                                            // bf16 [G4][KC]
constexpr size_t WS_BC    = WS_WC   + (size_t)G4 * KC * 2;                // f32  [G4]
constexpr size_t WS_WFC   = WS_BC   + (size_t)G4 * 4;                     // bf16 [OUTF][HIDDEN]
constexpr size_t WS_XBF   = WS_WFC  + (size_t)OUTF * HIDDEN * 2;          // bf16 [SEQ][BATCH][INPUT]
constexpr size_t WS_H     = WS_XBF  + (size_t)SEQ * BATCH * INPUT * 2;    // f32  [BATCH][HIDDEN]
constexpr size_t WS_C     = WS_H    + (size_t)BATCH * HIDDEN * 4;         // f32  [BATCH][HIDDEN]
constexpr size_t WS_HBFA  = WS_C    + (size_t)BATCH * HIDDEN * 4;         // bf16 [BATCH][HIDDEN]
constexpr size_t WS_HBFB  = WS_HBFA + (size_t)BATCH * HIDDEN * 2;        // bf16 [BATCH][HIDDEN]

// ---------------- types ----------------
typedef __attribute__((ext_vector_type(16))) __bf16 bf16x16;
typedef __attribute__((ext_vector_type(8)))  float  f32x8;
typedef int v4i_vs __attribute__((vector_size(16)));   // matches builtin param type
typedef __attribute__((address_space(1))) v4i_vs* gas_v4i;
typedef __attribute__((address_space(3))) v4i_vs* las_v4i;

#if __has_builtin(__builtin_amdgcn_global_load_async_to_lds_b128)
#define HAVE_ASYNC_LDS 1
#else
#define HAVE_ASYNC_LDS 0
#endif

__device__ __forceinline__ unsigned short f2bf(float f) {
  unsigned u = __float_as_uint(f);
  u += 0x7FFFu + ((u >> 16) & 1u);     // round-to-nearest-even
  return (unsigned short)(u >> 16);
}

__device__ __forceinline__ float sigm(float x) { return 1.0f / (1.0f + __expf(-x)); }

// 16-byte stage: GLOBAL_LOAD_ASYNC_TO_LDS_B128 (ASYNCcnt) or sync fallback.
__device__ __forceinline__ void stage16(const unsigned short* g, unsigned short* l) {
#if HAVE_ASYNC_LDS
  __builtin_amdgcn_global_load_async_to_lds_b128((gas_v4i)g, (las_v4i)l, 0, 0);
#else
  *reinterpret_cast<uint4*>(l) = *reinterpret_cast<const uint4*>(g);
#endif
}

__device__ __forceinline__ void wait_stage() {
#if HAVE_ASYNC_LDS
#if __has_builtin(__builtin_amdgcn_s_wait_asynccnt)
  __builtin_amdgcn_s_wait_asynccnt(0);
#else
  asm volatile("s_wait_asynccnt 0x0" ::: "memory");
#endif
#endif
}

// A/B fragment loader for v_wmma_f32_16x16x32_bf16 (wave32).
// A 16x32: lane l -> row (l&15); elems 0..7 = K[kb + 8*(l>>4) .. +7],
//          elems 8..15 = K[kb + 16 + 8*(l>>4) .. +7].  B (32x16) mirrors per column.
template <typename P>
__device__ __forceinline__ bf16x16 load_frag16(P row, int kbase, int half) {
  union { bf16x16 v; uint4 q[2]; } u;
  u.q[0] = *(const uint4*)(row + kbase + half * 8);
  u.q[1] = *(const uint4*)(row + kbase + 16 + half * 8);
  return u.v;
}

// ---------------- prep kernels ----------------
__global__ void prep_wc(const float* __restrict__ W_ih, const float* __restrict__ W_hh,
                        const float* __restrict__ b_ih, const float* __restrict__ b_hh,
                        unsigned short* __restrict__ Wc, float* __restrict__ bc) {
  const size_t total = (size_t)G4 * KC;
  for (size_t i = (size_t)blockIdx.x * blockDim.x + threadIdx.x; i < total;
       i += (size_t)gridDim.x * blockDim.x) {
    int g = (int)(i / KC), k = (int)(i % KC);
    float v = (k < INPUT) ? W_ih[(size_t)g * INPUT + k]
                          : W_hh[(size_t)g * HIDDEN + (k - INPUT)];
    Wc[i] = f2bf(v);
    if (i < (size_t)G4) bc[i] = b_ih[i] + b_hh[i];
  }
}

__global__ void prep_wfc(const float* __restrict__ W_fc, unsigned short* __restrict__ Wfc) {
  const size_t total = (size_t)OUTF * HIDDEN;
  for (size_t i = (size_t)blockIdx.x * blockDim.x + threadIdx.x; i < total;
       i += (size_t)gridDim.x * blockDim.x)
    Wfc[i] = f2bf(W_fc[i]);
}

// x (B, I, T) row-major -> xbf[t][b][i] bf16
__global__ void prep_x(const float* __restrict__ x, unsigned short* __restrict__ xbf) {
  const size_t total = (size_t)SEQ * BATCH * INPUT;
  for (size_t i = (size_t)blockIdx.x * blockDim.x + threadIdx.x; i < total;
       i += (size_t)gridDim.x * blockDim.x) {
    int t   = (int)(i / ((size_t)BATCH * INPUT));
    int rem = (int)(i % ((size_t)BATCH * INPUT));
    int b = rem / INPUT, ii = rem % INPUT;
    xbf[i] = f2bf(x[(size_t)b * INPUT * SEQ + (size_t)ii * SEQ + t]);
  }
}

__global__ void init_state(float* __restrict__ h, float* __restrict__ c,
                           unsigned short* __restrict__ hbfA, unsigned short* __restrict__ hbfB) {
  const size_t total = (size_t)BATCH * HIDDEN;
  for (size_t i = (size_t)blockIdx.x * blockDim.x + threadIdx.x; i < total;
       i += (size_t)gridDim.x * blockDim.x) {
    h[i] = 0.0f; c[i] = 0.0f; hbfA[i] = 0; hbfB[i] = 0;
  }
}

// ---------------- fused LSTM step ----------------
// 4-wave workgroup: 128-batch x 16-hidden tile, all 4 gates.
// Each wave: 32 batch rows (2 row-blocks) x 4 gates = 8 f32 accumulators.
// Gate weights double-buffered in LDS via async-to-LDS; A rows from global.
__global__ __launch_bounds__(128) void
lstm_step(const unsigned short* __restrict__ Wc,   // [G4][KC] bf16
          const float* __restrict__ bc,            // [G4]
          const unsigned short* __restrict__ xt,   // [BATCH][INPUT] bf16 (this timestep)
          const unsigned short* __restrict__ hin,  // [BATCH][HIDDEN] bf16
          unsigned short* __restrict__ hout,       // [BATCH][HIDDEN] bf16
          float* __restrict__ h, float* __restrict__ c) {
  __shared__ __align__(16) unsigned short smem[2][4][16][CK];  // 16 KB

  const int tid  = threadIdx.x;
  const int wave = tid >> 5;
  const int lane = tid & 31;
  const int half = lane >> 4;        // 0 | 1
  const int nsub = lane & 15;        // column / row-in-tile selector
  const int b0 = blockIdx.x * 128 + wave * 32;
  const int n0 = blockIdx.y * 16;

  const unsigned short* arow_x0 = xt  + (size_t)(b0 + nsub) * INPUT;
  const unsigned short* arow_h0 = hin + (size_t)(b0 + nsub) * HIDDEN;
  const unsigned short* arow_x1 = arow_x0 + (size_t)16 * INPUT;
  const unsigned short* arow_h1 = arow_h0 + (size_t)16 * HIDDEN;

  // cooperative weight staging: 512 x 16B segments per chunk, 4 per thread
  // seg s: gate = s>>7, row = (s&127)>>3, k16 = s&7
  auto stage_chunk = [&](int kb, int buf) {
#pragma unroll
    for (int j = 0; j < 4; ++j) {
      const int s   = tid + j * 128;
      const int g   = s >> 7;
      const int row = (s & 127) >> 3;
      const int k16 = s & 7;
      stage16(Wc + (size_t)(g * HIDDEN + n0 + row) * KC + kb + k16 * 8,
              &smem[buf][g][row][k16 * 8]);
    }
  };

  f32x8 acc[2][4] = {};

  stage_chunk(0, 0);
  wait_stage();
  __syncthreads();

  for (int ci = 0; ci < NCHUNK; ++ci) {
    const int cur = ci & 1;
    if (ci + 1 < NCHUNK) stage_chunk((ci + 1) * CK, cur ^ 1);

#pragma unroll
    for (int ks = 0; ks < 2; ++ks) {
      const int kglob = ci * CK + ks * 32;
      // all four gate B-fragments up front -> independent ds_loads pipeline
      bf16x16 bfr[4];
#pragma unroll
      for (int gi = 0; gi < 4; ++gi)
        bfr[gi] = load_frag16(&smem[cur][gi][nsub][0], ks * 32, half);

      bf16x16 a0, a1;
      if (kglob < INPUT) {
        a0 = load_frag16(arow_x0, kglob, half);
        a1 = load_frag16(arow_x1, kglob, half);
      } else {
        a0 = load_frag16(arow_h0, kglob - INPUT, half);
        a1 = load_frag16(arow_h1, kglob - INPUT, half);
      }
#pragma unroll
      for (int gi = 0; gi < 4; ++gi)
        acc[0][gi] = __builtin_amdgcn_wmma_f32_16x16x32_bf16(
            false, a0, false, bfr[gi], (short)0, acc[0][gi], false, false);
#pragma unroll
      for (int gi = 0; gi < 4; ++gi)
        acc[1][gi] = __builtin_amdgcn_wmma_f32_16x16x32_bf16(
            false, a1, false, bfr[gi], (short)0, acc[1][gi], false, false);
    }

    wait_stage();
    __syncthreads();   // chunk ci fully consumed; next buffer ready
  }

  const int nglob = n0 + nsub;
  const float bi = bc[nglob];
  const float bf = bc[HIDDEN + nglob];
  const float bg = bc[2 * HIDDEN + nglob];
  const float bo = bc[3 * HIDDEN + nglob];

#pragma unroll
  for (int rb = 0; rb < 2; ++rb) {
#pragma unroll
    for (int r = 0; r < 8; ++r) {
      const int m = rb * 16 + r + 8 * half;           // C/D layout: VGPR r, lane half
      const size_t idx = (size_t)(b0 + m) * HIDDEN + nglob;
      float ig = sigm(acc[rb][0][r] + bi);
      float fg = sigm(acc[rb][1][r] + bf);
      float gg = tanhf(acc[rb][2][r] + bg);
      float og = sigm(acc[rb][3][r] + bo);
      float cn = fg * c[idx] + ig * gg;
      float hn = og * tanhf(cn);
      c[idx] = cn;
      h[idx] = hn;
      hout[idx] = f2bf(hn);
    }
  }
}

// ---------------- final FC: out = h_last @ W_fc^T + b_fc ----------------
__global__ __launch_bounds__(32) void
fc_kernel(const unsigned short* __restrict__ hlast,   // [BATCH][HIDDEN] bf16
          const unsigned short* __restrict__ Wfc,     // [OUTF][HIDDEN] bf16
          const float* __restrict__ b_fc, float* __restrict__ out) {
  const int lane = threadIdx.x;
  const int half = lane >> 4;
  const int nsub = lane & 15;
  const int b0 = blockIdx.x * 16;
  const int n0 = blockIdx.y * 16;

  const unsigned short* arow = hlast + (size_t)(b0 + nsub) * HIDDEN;
  const unsigned short* brow = Wfc   + (size_t)(n0 + nsub) * HIDDEN;

  f32x8 acc = {};
  for (int kb = 0; kb < HIDDEN; kb += 32) {
    bf16x16 a = load_frag16(arow, kb, half);
    bf16x16 b = load_frag16(brow, kb, half);
    acc = __builtin_amdgcn_wmma_f32_16x16x32_bf16(
        false, a, false, b, (short)0, acc, false, false);
  }

  const float bias = b_fc[n0 + nsub];
#pragma unroll
  for (int r = 0; r < 8; ++r) {
    const int m = r + 8 * half;
    out[(size_t)(b0 + m) * OUTF + (n0 + nsub)] = acc[r] + bias;
  }
}

// ---------------- launcher ----------------
extern "C" void kernel_launch(void* const* d_in, const int* in_sizes, int n_in,
                              void* d_out, int out_size, void* d_ws, size_t ws_size,
                              hipStream_t stream) {
  const float* x    = (const float*)d_in[0];
  const float* W_ih = (const float*)d_in[1];
  const float* W_hh = (const float*)d_in[2];
  const float* b_ih = (const float*)d_in[3];
  const float* b_hh = (const float*)d_in[4];
  const float* W_fc = (const float*)d_in[5];
  const float* b_fc = (const float*)d_in[6];
  float* out = (float*)d_out;

  char* ws = (char*)d_ws;
  unsigned short* Wc   = (unsigned short*)(ws + WS_WC);
  float*          bc   = (float*)         (ws + WS_BC);
  unsigned short* Wfc  = (unsigned short*)(ws + WS_WFC);
  unsigned short* xbf  = (unsigned short*)(ws + WS_XBF);
  float*          h    = (float*)         (ws + WS_H);
  float*          c    = (float*)         (ws + WS_C);
  unsigned short* hbfA = (unsigned short*)(ws + WS_HBFA);
  unsigned short* hbfB = (unsigned short*)(ws + WS_HBFB);

  // prep
  prep_wc <<<4096, 256, 0, stream>>>(W_ih, W_hh, b_ih, b_hh, Wc, bc);
  prep_wfc<<<512,  256, 0, stream>>>(W_fc, Wfc);
  prep_x  <<<8192, 256, 0, stream>>>(x, xbf);
  init_state<<<2048, 256, 0, stream>>>(h, c, hbfA, hbfB);

  // recurrent chain: 255 dependent fused step-GEMMs (graph replay amortizes launches)
  dim3 grid(BATCH / 128, HIDDEN / 16);
  for (int t = 0; t < SEQ - 1; ++t) {
    const unsigned short* hin  = (t & 1) ? hbfB : hbfA;
    unsigned short*       hout = (t & 1) ? hbfA : hbfB;
    lstm_step<<<grid, 128, 0, stream>>>(Wc, bc,
                                        xbf + (size_t)t * BATCH * INPUT,
                                        hin, hout, h, c);
  }

  const unsigned short* hlast = ((SEQ - 2) & 1) ? hbfA : hbfB;
  dim3 gfc(BATCH / 16, OUTF / 16);
  fc_kernel<<<gfc, 32, 0, stream>>>(hlast, Wfc, b_fc, out);
}
